// Attention_1185410974116
// MI455X (gfx1250) — compile-verified, matrix operand order checked by reading, not run on hardware
//
#include <hip/hip_runtime.h>

// ---------------------------------------------------------------------------
// MI455X (gfx1250, wave32) implementation of the hybrid linear/softmax
// attention reference. All contractions use V_WMMA_F32_16X16X32_F16 with
// fp32 accumulation; operand tiles are staged through LDS (via the CDNA5
// async global->LDS path when the toolchain exposes it) and shared by the
// 8 waves of each 256-thread block.
// ---------------------------------------------------------------------------

typedef __attribute__((ext_vector_type(16))) _Float16 v16h;
typedef __attribute__((ext_vector_type(8)))  float    v8f;

constexpr int Bc  = 4;
constexpr int Sc  = 1024;
constexpr int DIM = 2048;
constexpr int Hc  = 32;
constexpr int HD  = 64;

// ---- CDNA5 async global->LDS support (guarded; falls back to b128 copies) --
#ifndef __has_builtin
#define __has_builtin(x) 0
#endif
#if __has_builtin(__builtin_amdgcn_global_load_async_to_lds_b128)
#define USE_ASYNC_LDS 1
#else
#define USE_ASYNC_LDS 0
#endif

#if USE_ASYNC_LDS
// Probe-verified parameter type on this toolchain: generic `int4*`
// ("__attribute__((__vector_size__(4*sizeof(int)))) int *").
typedef int v4i_ __attribute__((__vector_size__(16)));
static __device__ __forceinline__ void async_g2l_b128(const void* g, void* l) {
  __builtin_amdgcn_global_load_async_to_lds_b128((v4i_*)g, (v4i_*)l, 0, 0);
}
static __device__ __forceinline__ void wait_async0() {
#if __has_builtin(__builtin_amdgcn_s_wait_asynccnt)
  __builtin_amdgcn_s_wait_asynccnt(0);
#else
  asm volatile("s_wait_asynccnt 0x0" ::: "memory");
#endif
}
#endif

// Stage 16B per call: async DMA when available, else load/store through VGPRs.
static __device__ __forceinline__ void stage16(const float* g, float* l) {
#if USE_ASYNC_LDS
  async_g2l_b128(g, l);
#else
  *(float4*)l = *(const float4*)g;
#endif
}
static __device__ __forceinline__ void stage_fence() {
#if USE_ASYNC_LDS
  wait_async0();
#endif
}

// ---- WMMA fragment helpers (CDNA5 wave32 layouts, CDNA5 ISA) ----
// A 16x32 f16: lane = row (0..15, both halves), lane-half selects k-groups:
//   half h, j<8  -> k = 8h + j ; j>=8 -> k = 16 + 8h + (j-8)
// C/D 16x16 f32: lane&15 = col, VGPR r = row (lanes 16-31: row r+8).

static __device__ __forceinline__ v16h pack16(float4 f0, float4 f1,
                                              float4 f2, float4 f3) {
  v16h a;
  a[0]=(_Float16)f0.x;  a[1]=(_Float16)f0.y;  a[2]=(_Float16)f0.z;  a[3]=(_Float16)f0.w;
  a[4]=(_Float16)f1.x;  a[5]=(_Float16)f1.y;  a[6]=(_Float16)f1.z;  a[7]=(_Float16)f1.w;
  a[8]=(_Float16)f2.x;  a[9]=(_Float16)f2.y;  a[10]=(_Float16)f2.z; a[11]=(_Float16)f2.w;
  a[12]=(_Float16)f3.x; a[13]=(_Float16)f3.y; a[14]=(_Float16)f3.z; a[15]=(_Float16)f3.w;
  return a;
}

// p already points at (row_base + half*8); contiguous k runs of 8 at +0 and +16.
static __device__ __forceinline__ v16h frag_rowmajor(const float* p) {
  return pack16(*(const float4*)(p),      *(const float4*)(p + 4),
                *(const float4*)(p + 16), *(const float4*)(p + 20));
}

static __device__ __forceinline__ int kpat(int half, int j) {
  return (j < 8) ? (half * 8 + j) : (16 + half * 8 + (j - 8));
}

// Strided fragment gather (k dimension has stride `stride` floats in LDS).
static __device__ __forceinline__ v16h frag_strided(const float* p, int half,
                                                    int stride) {
  v16h a;
#pragma unroll
  for (int j = 0; j < 16; ++j) a[j] = (_Float16)p[kpat(half, j) * stride];
  return a;
}

static __device__ __forceinline__ v8f wmma_f16(v16h a, v16h b, v8f c) {
  return __builtin_amdgcn_wmma_f32_16x16x32_f16(false, a, false, b,
                                                (short)0, c, false, false);
}

static __device__ __forceinline__ float elu1f(float x) {
  return x > 0.f ? x + 1.f : __expf(fminf(x, 0.f));
}
static __device__ __forceinline__ float sigmoidf_(float x) {
  return 1.f / (1.f + __expf(-x));
}

// ---------------------------------------------------------------------------
// Generic Y[m,n] (+)= sum_k X[m,k] * W[n,k].  Block = 8 waves = 128 M-rows x
// 16 N-cols; the W tile (16x32) is double-buffered in LDS: the async copy of
// tile i+1 overlaps the WMMA on tile i.
// ---------------------------------------------------------------------------
template <int ACC>
__global__ void gemm_nt_k(float* __restrict__ Y, const float* __restrict__ X,
                          const float* __restrict__ W,
                          int ldx, int ldw, int ldy, int Kt) {
  __shared__ float Bs[2][16 * 32];
  const int lane = threadIdx.x & 31, wave = threadIdx.x >> 5;
  const int half = lane >> 4, mr = lane & 15;
  const int n0 = blockIdx.x * 16;
  const int m0 = blockIdx.y * 128 + wave * 16;
  const float* xrow = X + (size_t)(m0 + mr) * ldx;

  auto stage_tile = [&](int k0, int buf) {
    if (threadIdx.x < 128) {
      const int row = threadIdx.x >> 3;
      const int ko  = (threadIdx.x & 7) * 4;
      stage16(W + (size_t)(n0 + row) * ldw + k0 + ko, &Bs[buf][row * 32 + ko]);
    }
  };

  stage_tile(0, 0);
  v8f c = {};
  for (int k0 = 0; k0 < Kt; k0 += 32) {
    const int buf = (k0 >> 5) & 1;
    stage_fence();                      // drain this wave's async copies
    __syncthreads();                    // tile visible block-wide
    if (k0 + 32 < Kt) stage_tile(k0 + 32, buf ^ 1);   // prefetch next tile
    const float* ap = xrow + k0 + half * 8;
    __builtin_prefetch(ap + 32, 0, 0);  // global_prefetch_b8 on A stream
    v16h a = frag_rowmajor(ap);
    v16h b = frag_rowmajor(&Bs[buf][mr * 32 + half * 8]);
    c = wmma_f16(a, b, c);
    __syncthreads();                    // reads done before buf is re-filled
  }
#pragma unroll
  for (int r = 0; r < 8; ++r) {
    size_t off = (size_t)(m0 + half * 8 + r) * ldy + n0 + mr;
    if (ACC) Y[off] += c[r]; else Y[off] = c[r];
  }
}

// ---------------------------------------------------------------------------
// RoPE + elu1 + head-major transposes (elementwise).
// ---------------------------------------------------------------------------
__global__ void rope_elu_k(const float* __restrict__ XQb, const float* __restrict__ XKb,
                           const float* __restrict__ XVb, const float* __restrict__ CK,
                           const float* __restrict__ FC,  const float* __restrict__ FS,
                           float* __restrict__ QRb, float* __restrict__ KRb,
                           float* __restrict__ SQb, float* __restrict__ VTb,
                           float* __restrict__ SKb) {
  int i = blockIdx.x * 256 + threadIdx.x;          // B*S*H*(HD/2) pairs
  if (i >= Bc * Sc * Hc * (HD / 2)) return;
  int j = i & 31; int rem = i >> 5;
  int h = rem & (Hc - 1); rem >>= 5;
  int s = rem & (Sc - 1); int b = rem >> 10;
  int d0 = j * 2;
  size_t xo = ((size_t)(b * Sc + s) * DIM) + h * HD + d0;        // [B,S,H,HD]
  size_t to = (((size_t)(b * Hc + h) * Sc + s) * HD) + d0;       // [B,H,S,HD]
  float cz = FC[s * 32 + j], sn = FS[s * 32 + j];
  float qa = XQb[xo], qb = XQb[xo + 1];
  QRb[to] = qa * cz - qb * sn;  QRb[to + 1] = qa * sn + qb * cz;
  float ka = XKb[xo], kb2 = XKb[xo + 1];
  KRb[to] = ka * cz - kb2 * sn; KRb[to + 1] = ka * sn + kb2 * cz;
  SQb[to] = elu1f(qa);          SQb[to + 1] = elu1f(qb);
  VTb[to] = XVb[xo];            VTb[to + 1] = XVb[xo + 1];
  float ca = CK[xo], cb = CK[xo + 1];
  SKb[to] = elu1f(ca);          SKb[to + 1] = elu1f(cb);
}

// ---------------------------------------------------------------------------
// q_scores -> sigmoid -> query_f_v.  One wave per (b,h,s) row, shfl reduction.
// ---------------------------------------------------------------------------
__global__ void qscore_qfv_k(float* __restrict__ QFV, const float* __restrict__ PQ,
                             const float* __restrict__ CK, const float* __restrict__ CV) {
  int idx  = blockIdx.x * 8 + (threadIdx.x >> 5);
  int lane = threadIdx.x & 31;
  int s = idx & (Sc - 1); int bh = idx >> 10;
  int b = bh >> 5; int h = bh & (Hc - 1);
  int d = lane * 2;
  size_t co = ((size_t)(b * Sc + s) * Hc + h) * HD + d;
  size_t po = ((size_t)b * Hc + h) * HD + d;
  float2 pq2 = *(const float2*)(PQ + po);
  float2 ck2 = *(const float2*)(CK + co);
  float part = pq2.x * ck2.x + pq2.y * ck2.y;
#pragma unroll
  for (int m = 16; m >= 1; m >>= 1) part += __shfl_xor(part, m, 32);
  float sig = sigmoidf_(part * 0.125f);            // 1/sqrt(64)
  float2 cv2 = *(const float2*)(CV + co);
  size_t qo = ((size_t)bh * Sc + s) * HD + d;
  QFV[qo] = cv2.x * sig; QFV[qo + 1] = cv2.y * sig;
}

// nrm_t[b,h,d] = norm_term + sum_s sigma_k
__global__ void nrm_sum_k(float* __restrict__ NT, const float* __restrict__ NTin,
                          const float* __restrict__ SKb) {
  int i = blockIdx.x * 256 + threadIdx.x;
  if (i >= Bc * Hc * HD) return;
  int bh = i >> 6; int d = i & 63;
  const float* p = SKb + (size_t)bh * Sc * HD + d;
  float acc = NTin[i];
  for (int s = 0; s < Sc; ++s) acc += p[(size_t)s * HD];
  NT[i] = acc;
}

// ---------------------------------------------------------------------------
// mem[b,h] = base + sk^T @ rhs   (64x64 = sum_s outer products), WMMA over
// LDS-staged [32 x 64] chunks of both operands. One block per (b,h).
// ---------------------------------------------------------------------------
__global__ void mem_update_k(float* __restrict__ dst, const float* __restrict__ base,
                             const float* __restrict__ SKb, const float* __restrict__ RHS) {
  __shared__ float As[32 * 64];
  __shared__ float Bs[32 * 64];
  const int lane = threadIdx.x & 31, wave = threadIdx.x >> 5;
  const int half = lane >> 4, mr = lane & 15;
  const int bh = blockIdx.x;
  const float* ab = SKb + (size_t)bh * Sc * HD;
  const float* bb = RHS + (size_t)bh * Sc * HD;
  const int t0 = wave * 2, t1 = wave * 2 + 1;
  const int mt0 = t0 >> 2, nt0 = t0 & 3, mt1 = t1 >> 2, nt1 = t1 & 3;
  v8f c0 = {}, c1 = {};
  for (int s0 = 0; s0 < Sc; s0 += 32) {
    __syncthreads();
    {
      int r = threadIdx.x >> 3, co = (threadIdx.x & 7) * 8;
      const float* ap = ab + (size_t)(s0 + r) * HD + co;
      stage16(ap,     &As[r * 64 + co]);
      stage16(ap + 4, &As[r * 64 + co + 4]);
      const float* bp = bb + (size_t)(s0 + r) * HD + co;
      stage16(bp,     &Bs[r * 64 + co]);
      stage16(bp + 4, &Bs[r * 64 + co + 4]);
      stage_fence();
    }
    __syncthreads();
    c0 = wmma_f16(frag_strided(&As[mt0 * 16 + mr], half, 64),
                  frag_strided(&Bs[nt0 * 16 + mr], half, 64), c0);
    c1 = wmma_f16(frag_strided(&As[mt1 * 16 + mr], half, 64),
                  frag_strided(&Bs[nt1 * 16 + mr], half, 64), c1);
  }
#pragma unroll
  for (int r = 0; r < 8; ++r) {
    { int d = mt0 * 16 + half * 8 + r, e = nt0 * 16 + mr;
      size_t off = (size_t)bh * HD * HD + d * HD + e;
      dst[off] = base[off] + c0[r]; }
    { int d = mt1 * 16 + half * 8 + r, e = nt1 * 16 + mr;
      size_t off = (size_t)bh * HD * HD + d * HD + e;
      dst[off] = base[off] + c1[r]; }
  }
}

// dst[b,h,s,e] = sum_d SQ[b,h,s,d] * MEM[b,h,d,e]  (raw, no denom yet)
__global__ void apply_mem_k(float* __restrict__ dst, const float* __restrict__ SQb,
                            const float* __restrict__ MEM) {
  __shared__ float Ms[64 * 64];
  const int lane = threadIdx.x & 31, wave = threadIdx.x >> 5;
  const int half = lane >> 4, mr = lane & 15;
  const int bh = blockIdx.x;
  {
    const float* mp = MEM + (size_t)bh * HD * HD + threadIdx.x * 16;
#pragma unroll
    for (int i = 0; i < 4; ++i)
      stage16(mp + i * 4, &Ms[threadIdx.x * 16 + i * 4]);
    stage_fence();
  }
  __syncthreads();
  const int q0 = blockIdx.y * 128 + wave * 16;
  const float* sp = SQb + ((size_t)bh * Sc + q0 + mr) * HD;
  v16h a0 = frag_rowmajor(sp + half * 8);
  v16h a1 = frag_rowmajor(sp + 32 + half * 8);
#pragma unroll
  for (int e0 = 0; e0 < 4; ++e0) {
    v8f c = {};
    c = wmma_f16(a0, frag_strided(&Ms[e0 * 16 + mr], half, 64), c);
    c = wmma_f16(a1, frag_strided(&Ms[32 * 64 + e0 * 16 + mr], half, 64), c);
#pragma unroll
    for (int r = 0; r < 8; ++r) {
      size_t off = ((size_t)bh * Sc + q0 + half * 8 + r) * HD + e0 * 16 + mr;
      dst[off] = c[r];
    }
  }
}

// ---------------------------------------------------------------------------
// Causal softmax attention, flash-style. Block = (b,h, 128 q rows), 8 waves of
// 16 q-rows; K/V key-chunks [32 x 64] staged in LDS and shared by all waves.
// Scores and P.V both on WMMA; exp tile round-trips wave-private LDS (same-
// wave DS is in-order, so no block barrier is needed for it).
// ---------------------------------------------------------------------------
__global__ void flash_attn_k(float* __restrict__ attn, const float* __restrict__ QRb,
                             const float* __restrict__ KRb, const float* __restrict__ VTb) {
  __shared__ float Ks[32 * 64];
  __shared__ float Vs[32 * 64];
  __shared__ float Es[8][16 * 32];
  __shared__ float Ls[8][16];
  const int lane = threadIdx.x & 31, wave = threadIdx.x >> 5;
  const int half = lane >> 4, mr = lane & 15;
  const int bh = blockIdx.x;
  const int q0b = blockIdx.y * 128;
  const int q0 = q0b + wave * 16;
  const float* qp = QRb + ((size_t)bh * Sc + (q0 + mr)) * HD;
  v16h aq0 = frag_rowmajor(qp + half * 8);
  v16h aq1 = frag_rowmajor(qp + 32 + half * 8);
  v8f acc0 = {}, acc1 = {}, acc2 = {}, acc3 = {};
  float l_i = 0.f;
  const float* kb = KRb + (size_t)bh * Sc * HD;
  const float* vb = VTb + (size_t)bh * Sc * HD;
  const int smax = q0b + 128;                       // block-uniform causal bound
  for (int s0 = 0; s0 < smax; s0 += 32) {
    __syncthreads();
    {
      int r = threadIdx.x >> 3, co = (threadIdx.x & 7) * 8;
      const float* kp = kb + (size_t)(s0 + r) * HD + co;
      stage16(kp,     &Ks[r * 64 + co]);
      stage16(kp + 4, &Ks[r * 64 + co + 4]);
      const float* vp = vb + (size_t)(s0 + r) * HD + co;
      stage16(vp,     &Vs[r * 64 + co]);
      stage16(vp + 4, &Vs[r * 64 + co + 4]);
      __builtin_prefetch(kp + 32 * HD, 0, 0);       // next key chunk
      stage_fence();
    }
    __syncthreads();
    v8f sc0 = {}, sc1 = {};
    sc0 = wmma_f16(aq0, frag_rowmajor(&Ks[mr * 64 + half * 8]), sc0);
    sc0 = wmma_f16(aq1, frag_rowmajor(&Ks[mr * 64 + 32 + half * 8]), sc0);
    sc1 = wmma_f16(aq0, frag_rowmajor(&Ks[(16 + mr) * 64 + half * 8]), sc1);
    sc1 = wmma_f16(aq1, frag_rowmajor(&Ks[(16 + mr) * 64 + 32 + half * 8]), sc1);
#pragma unroll
    for (int r = 0; r < 8; ++r) {
      int row  = q0 + half * 8 + r;
      int key0 = s0 + mr;
      float e0v = (key0      <= row) ? __expf(sc0[r] * 0.125f) : 0.f;
      float e1v = (key0 + 16 <= row) ? __expf(sc1[r] * 0.125f) : 0.f;
      Es[wave][(half * 8 + r) * 32 + mr]      = e0v;
      Es[wave][(half * 8 + r) * 32 + 16 + mr] = e1v;
    }
    // wave-private LDS round trip: same-wave DS ops are in-order (DScnt).
    {
      float rs = 0.f;
#pragma unroll
      for (int j = 0; j < 8; ++j) {
        float4 e4 = *(const float4*)&Es[wave][mr * 32 + j * 4];
        rs += e4.x + e4.y + e4.z + e4.w;
      }
      l_i += rs;
    }
    v16h ap = frag_rowmajor(&Es[wave][mr * 32 + half * 8]);
    acc0 = wmma_f16(ap, frag_strided(&Vs[mr],      half, 64), acc0);
    acc1 = wmma_f16(ap, frag_strided(&Vs[16 + mr], half, 64), acc1);
    acc2 = wmma_f16(ap, frag_strided(&Vs[32 + mr], half, 64), acc2);
    acc3 = wmma_f16(ap, frag_strided(&Vs[48 + mr], half, 64), acc3);
  }
  if (lane < 16) Ls[wave][mr] = l_i;                // same-wave DS is in-order
#pragma unroll
  for (int r = 0; r < 8; ++r) {
    float linv = 1.f / Ls[wave][half * 8 + r];
    size_t ro = ((size_t)bh * Sc + (q0 + half * 8 + r)) * HD;
    attn[ro +      mr] = acc0[r] * linv;
    attn[ro + 16 + mr] = acc1[r] * linv;
    attn[ro + 32 + mr] = acc2[r] * linv;
    attn[ro + 48 + mr] = acc3[r] * linv;
  }
}

// ---------------------------------------------------------------------------
// Denominator, qmem gate, head gate, final mix.  Wave per (b,h,s) row.
// ---------------------------------------------------------------------------
__global__ void combine_k(float* __restrict__ COMB, const float* __restrict__ MOUTr,
                          const float* __restrict__ QMOUTr, const float* __restrict__ SQb,
                          const float* __restrict__ NT, const float* __restrict__ ATTNb,
                          const float* __restrict__ GATE, const float* __restrict__ GW,
                          const float* __restrict__ GB) {
  int idx  = blockIdx.x * 8 + (threadIdx.x >> 5);
  int lane = threadIdx.x & 31;
  int s = idx & (Sc - 1); int bh = idx >> 10;
  int b = bh >> 5; int h = bh & (Hc - 1);
  int d = lane * 2;
  size_t off = ((size_t)bh * Sc + s) * HD + d;
  float2 sq2 = *(const float2*)(SQb + off);
  float2 nt2 = *(const float2*)(NT + (size_t)bh * HD + d);
  float part = sq2.x * nt2.x + sq2.y * nt2.y;
#pragma unroll
  for (int m = 16; m >= 1; m >>= 1) part += __shfl_xor(part, m, 32);
  float inv = 1.f / part;
  float2 mo  = *(const float2*)(MOUTr + off);
  float2 qmo = *(const float2*)(QMOUTr + off);
  mo.x *= inv; mo.y *= inv; qmo.x *= inv; qmo.y *= inv;
  float gp = qmo.x * GW[d] + qmo.y * GW[d + 1];
#pragma unroll
  for (int m = 16; m >= 1; m >>= 1) gp += __shfl_xor(gp, m, 32);
  float gq = sigmoidf_(gp + GB[0]);
  float g  = sigmoidf_(GATE[h]);
  float2 at = *(const float2*)(ATTNb + off);
  float ox = g * (mo.x + gq * qmo.x) + (1.f - g) * at.x;
  float oy = g * (mo.y + gq * qmo.y) + (1.f - g) * at.y;
  size_t co = ((size_t)(b * Sc + s)) * DIM + h * HD + d;
  COMB[co] = ox; COMB[co + 1] = oy;
}

// ---- small utility kernels -------------------------------------------------
__global__ void zero_f32_k(float* p, int n) {
  int i = blockIdx.x * 256 + threadIdx.x;
  if (i < n) p[i] = 0.f;
}
__global__ void pad16to32_k(float* __restrict__ dst, const float* __restrict__ src,
                            int rows) {
  int i = blockIdx.x * 256 + threadIdx.x;
  if (i < rows * 32) {
    int r = i >> 5, c = i & 31;
    dst[i] = (c < 16) ? src[r * 16 + c] : 0.f;
  }
}

// ---------------------------------------------------------------------------
extern "C" void kernel_launch(void* const* d_in, const int* in_sizes, int n_in,
                              void* d_out, int out_size, void* d_ws, size_t ws_size,
                              hipStream_t stream) {
  (void)in_sizes; (void)n_in; (void)out_size; (void)ws_size;
  const float* x    = (const float*)d_in[0];
  const float* pq   = (const float*)d_in[1];
  const float* ck   = (const float*)d_in[2];
  const float* cv   = (const float*)d_in[3];
  const float* memL = (const float*)d_in[4];
  const float* memQ = (const float*)d_in[5];
  const float* nrm  = (const float*)d_in[6];
  const float* fc   = (const float*)d_in[7];
  const float* fs   = (const float*)d_in[8];
  const float* wq   = (const float*)d_in[10];
  const float* wk   = (const float*)d_in[11];
  const float* wv   = (const float*)d_in[12];
  const float* wo   = (const float*)d_in[13];
  const float* lq1  = (const float*)d_in[14];
  const float* lq2  = (const float*)d_in[15];
  const float* lk1  = (const float*)d_in[16];
  const float* lk2  = (const float*)d_in[17];
  const float* lv1  = (const float*)d_in[18];
  const float* lv2  = (const float*)d_in[19];
  const float* lo1  = (const float*)d_in[20];
  const float* lo2  = (const float*)d_in[21];
  const float* gate = (const float*)d_in[22];
  const float* gw   = (const float*)d_in[23];
  const float* gb   = (const float*)d_in[24];
  float* out = (float*)d_out;
  float* ws  = (float*)d_ws;

  constexpr size_t NE  = (size_t)Bc * Sc * DIM;          // 8,388,608
  constexpr size_t TSZ = (size_t)Bc * Sc * 32;           // 131,072 (padded T)
  constexpr size_t LSZ = (size_t)DIM * 32;               // 65,536 (padded L2)
  constexpr size_t MSZ = (size_t)Bc * Hc * HD * HD;      // 524,288

  float* XQ = ws;               float* XK = ws + NE;      float* XV = ws + 2*NE;
  float* QR = ws + 3*NE;        float* KR = ws + 4*NE;    float* SQ = ws + 5*NE;
  float* VT = ws + 6*NE;        float* SK = ws + 7*NE;    float* AT = ws + 8*NE;
  float* TQ = ws + 9*NE;        float* TK = TQ + TSZ;     float* TV = TK + TSZ;
  float* TO = TV + TSZ;
  float* L2Q = TO + TSZ;  float* L2K = L2Q + LSZ; float* L2V = L2K + LSZ;
  float* L2O = L2V + LSZ;
  float* MLb = L2O + LSZ; float* MQb = MLb + MSZ; float* NT = MQb + MSZ;
  // buffer reuse (strictly stream-ordered):
  float* QFV   = XV;   // xv consumed by rope_elu before qscore writes QFV
  float* MOUT  = XQ;   // xq consumed by rope_elu before apply_mem writes
  float* QMOUT = XK;
  float* COMB  = QR;   // qr consumed by flash_attn before combine writes

  dim3 blk(256);
  // 0) zero padded-T region (TQ..TO contiguous) + pad low-rank second factors
  zero_f32_k<<<dim3((4 * (int)TSZ + 255) / 256), blk, 0, stream>>>(TQ, 4 * (int)TSZ);
  pad16to32_k<<<dim3(((int)LSZ + 255) / 256), blk, 0, stream>>>(L2Q, lq2, DIM);
  pad16to32_k<<<dim3(((int)LSZ + 255) / 256), blk, 0, stream>>>(L2K, lk2, DIM);
  pad16to32_k<<<dim3(((int)LSZ + 255) / 256), blk, 0, stream>>>(L2V, lv2, DIM);
  pad16to32_k<<<dim3(((int)LSZ + 255) / 256), blk, 0, stream>>>(L2O, lo2, DIM);

  // 1) LoRA stage 1: T = x @ l1^T   (M=4096, N=16, K=2048)
  gemm_nt_k<0><<<dim3(1, 32), blk, 0, stream>>>(TQ, x, lq1, DIM, DIM, 32, DIM);
  gemm_nt_k<0><<<dim3(1, 32), blk, 0, stream>>>(TK, x, lk1, DIM, DIM, 32, DIM);
  gemm_nt_k<0><<<dim3(1, 32), blk, 0, stream>>>(TV, x, lv1, DIM, DIM, 32, DIM);
  // 2) main projections: X* = x @ W^T  (M=4096, N=2048, K=2048)
  gemm_nt_k<0><<<dim3(128, 32), blk, 0, stream>>>(XQ, x, wq, DIM, DIM, DIM, DIM);
  gemm_nt_k<0><<<dim3(128, 32), blk, 0, stream>>>(XK, x, wk, DIM, DIM, DIM, DIM);
  gemm_nt_k<0><<<dim3(128, 32), blk, 0, stream>>>(XV, x, wv, DIM, DIM, DIM, DIM);
  // 3) LoRA stage 2: X* += T @ L2p^T  (K=32 padded)
  gemm_nt_k<1><<<dim3(128, 32), blk, 0, stream>>>(XQ, TQ, L2Q, 32, 32, DIM, 32);
  gemm_nt_k<1><<<dim3(128, 32), blk, 0, stream>>>(XK, TK, L2K, 32, 32, DIM, 32);
  gemm_nt_k<1><<<dim3(128, 32), blk, 0, stream>>>(XV, TV, L2V, 32, 32, DIM, 32);

  // 4) RoPE / elu1 / transposes
  rope_elu_k<<<dim3(Bc * Sc * Hc * (HD / 2) / 256), blk, 0, stream>>>(
      XQ, XK, XV, ck, fc, fs, QR, KR, SQ, VT, SK);

  // 5) prompt-query gating of cached V
  qscore_qfv_k<<<dim3(Bc * Hc * Sc / 8), blk, 0, stream>>>(QFV, pq, ck, cv);
  // 6) norm term
  nrm_sum_k<<<dim3(Bc * Hc * HD / 256), blk, 0, stream>>>(NT, nrm, SK);
  // 7) memory updates (per-head 64x64)
  mem_update_k<<<dim3(Bc * Hc), blk, 0, stream>>>(MLb, memL, SK, VT);
  mem_update_k<<<dim3(Bc * Hc), blk, 0, stream>>>(MQb, memQ, SK, QFV);
  // 8) apply memories (raw, denom applied in combine)
  apply_mem_k<<<dim3(Bc * Hc, Sc / 128), blk, 0, stream>>>(MOUT, SQ, MLb);
  apply_mem_k<<<dim3(Bc * Hc, Sc / 128), blk, 0, stream>>>(QMOUT, SQ, MQb);
  // 9) softmax attention
  flash_attn_k<<<dim3(Bc * Hc, Sc / 128), blk, 0, stream>>>(AT, QR, KR, VT);
  // 10) gating + mix back to [B,S,DIM]
  combine_k<<<dim3(Bc * Hc * Sc / 8), blk, 0, stream>>>(COMB, MOUT, QMOUT, SQ, NT,
                                                        AT, gate, gw, gb);
  // 11) output projection + LoRA
  gemm_nt_k<0><<<dim3(1, 32), blk, 0, stream>>>(TO, COMB, lo1, DIM, DIM, 32, DIM);
  gemm_nt_k<0><<<dim3(128, 32), blk, 0, stream>>>(out, COMB, wo, DIM, DIM, DIM, DIM);
  gemm_nt_k<1><<<dim3(128, 32), blk, 0, stream>>>(out, TO, L2O, 32, 32, DIM, 32);
}
